// GCNNet_solvent_graph_66391604462259
// MI455X (gfx1250) — matrix operand order repaired and checked by reading, not compiled, not run on hardware
//
#include <hip/hip_runtime.h>
#include <hip/hip_bf16.h>

#define NN 50000
#define NE 800000
#define NG 500

typedef __attribute__((ext_vector_type(16))) __bf16 v16bf;
typedef __attribute__((ext_vector_type(8)))  float  v8f;

#define MAX_K   1024
#define LDS_STRIDE (MAX_K + 16)   // halves; pad keeps 32B alignment + breaks bank clash

// ---------------------------------------------------------------------------
// Wave-tile GEMM: C[M,N] = relu?(A[M,K] @ B[K,N] + bias)
// Block = 256 threads (8 waves). The block's 16-column B tile (K x 16) is
// staged once in LDS as bf16, [n][k] layout. Each wave computes a 32x16
// output tile: two v_wmma_f32_16x16x32_bf16 per K-step sharing one B fetch
// (two ds_load_b128 per lane). A rows stream from global as b128 + cvt.
// Row indices are CLAMPED (not predicated) so the K-loop is branch-free and
// EXEC stays all-1s (WMMA requirement); OOB rows are masked at the store.
// VGPR layouts per CDNA5 ISA 7.12.2 (wave32).
// ---------------------------------------------------------------------------
__global__ __launch_bounds__(256)
void gemm_wmma_bf16(const float* __restrict__ A, const float* __restrict__ B,
                    const float* __restrict__ bias, float* __restrict__ C,
                    int M, int K, int N, int relu)
{
    __shared__ __bf16 Bs[16 * LDS_STRIDE];

    const int tid   = threadIdx.x;
    const int lane  = tid & 31;
    const int wave  = tid >> 5;
    const int lo    = lane & 15;
    const int hi    = lane >> 4;
    const int nBase = blockIdx.x * 16;
    const int mBase = (blockIdx.y * 8 + wave) * 32;

    // --- cooperative stage of B tile: Bs[n][k] = (bf16) B[k][nBase+n]
    for (int idx = tid; idx < K * 16; idx += 256) {
        int n = idx & 15;
        int k = idx >> 4;
        Bs[n * LDS_STRIDE + k] = (__bf16)B[(size_t)k * N + (nBase + n)];
    }
    __syncthreads();

    if (mBase < M) {                       // wave-uniform: EXEC all-1s inside
        v8f acc0 = {};
        v8f acc1 = {};
        // clamp rows: OOB lanes read row M-1 (valid memory), results masked at store
        const int m0 = min(mBase + lo,      M - 1);
        const int m1 = min(mBase + 16 + lo, M - 1);
        const float* arow0 = A + (size_t)m0 * K;
        const float* arow1 = A + (size_t)m1 * K;
        const __bf16* bs   = Bs + lo * LDS_STRIDE;

        for (int k0 = 0; k0 < K; k0 += 32) {
            v16bf av0, av1, bv;
            {
                const float* ap = arow0 + k0 + hi * 8;
                #pragma unroll
                for (int i = 0; i < 8; ++i) {
                    av0[i]     = (__bf16)ap[i];        // K = k0 + hi*8 + i
                    av0[8 + i] = (__bf16)ap[16 + i];   // K = k0 + hi*8 + 16 + i
                }
            }
            {
                const float* ap = arow1 + k0 + hi * 8;
                #pragma unroll
                for (int i = 0; i < 8; ++i) {
                    av1[i]     = (__bf16)ap[i];
                    av1[8 + i] = (__bf16)ap[16 + i];
                }
            }
            // B column lo, K = k0 + hi*8 + [0..7] and k0 + 16 + hi*8 + [0..7]  (16B-aligned)
            {
                typedef __attribute__((ext_vector_type(8))) __bf16 v8bf;
                v8bf b0 = *(const v8bf*)(bs + k0 + hi * 8);
                v8bf b1 = *(const v8bf*)(bs + k0 + 16 + hi * 8);
                bv = __builtin_shufflevector(b0, b1, 0,1,2,3,4,5,6,7,8,9,10,11,12,13,14,15);
            }

            acc0 = __builtin_amdgcn_wmma_f32_16x16x32_bf16(
                       false, av0, false, bv, (short)0, acc0, false, false);
            acc1 = __builtin_amdgcn_wmma_f32_16x16x32_bf16(
                       false, av1, false, bv, (short)0, acc1, false, false);
        }

        const int   n  = nBase + lo;        // N is a multiple of 16 for all calls
        const float bb = bias ? bias[n] : 0.0f;
        #pragma unroll
        for (int r = 0; r < 8; ++r) {
            int mr = mBase + hi * 8 + r;
            if (mr < M) {
                float v = acc0[r] + bb;
                if (relu) v = fmaxf(v, 0.0f);
                C[(size_t)mr * N + n] = v;
            }
        }
        #pragma unroll
        for (int r = 0; r < 8; ++r) {
            int mr = mBase + 16 + hi * 8 + r;
            if (mr < M) {
                float v = acc1[r] + bb;
                if (relu) v = fmaxf(v, 0.0f);
                C[(size_t)mr * N + n] = v;
            }
        }
    }
}

// ---------------------------------------------------------------------------
// Small utility / message-passing kernels
// ---------------------------------------------------------------------------
__global__ void fill_f32(float* __restrict__ p, float v, long long n) {
    long long i = (long long)blockIdx.x * blockDim.x + threadIdx.x;
    if (i < n) p[i] = v;
}

__global__ void deg_accum(const int* __restrict__ col, float* __restrict__ deg, int E) {
    int e = blockIdx.x * blockDim.x + threadIdx.x;
    if (e < E) atomicAdd(&deg[col[e]], 1.0f);
}

__global__ void rsqrt_ip(float* __restrict__ d, int n) {
    int i = blockIdx.x * blockDim.x + threadIdx.x;
    if (i < n) d[i] = rsqrtf(d[i]);   // deg >= 1 always (self loop)
}

// agg[i] = T[i] * dinv[i]^2   (the self-loop term seeds the accumulator)
__global__ void self_loop_init(const float* __restrict__ T, const float* __restrict__ dinv,
                               float* __restrict__ agg, int M, int d)
{
    int chunk = d >> 2;
    long long tid = (long long)blockIdx.x * blockDim.x + threadIdx.x;
    if (tid >= (long long)M * chunk) return;
    int i = (int)(tid / chunk);
    int c = (int)(tid % chunk) << 2;
    float w = dinv[i] * dinv[i];
    float4 v = *(const float4*)(T + (size_t)i * d + c);
    v.x *= w; v.y *= w; v.z *= w; v.w *= w;
    *(float4*)(agg + (size_t)i * d + c) = v;
}

// agg[col] += T[row] * dinv[row]*dinv[col]  over all edges (float4 per thread)
__global__ void edge_scatter(const float* __restrict__ T,
                             const int* __restrict__ rowI, const int* __restrict__ colI,
                             const float* __restrict__ dinv, float* __restrict__ agg,
                             int E, int d)
{
    int chunk = d >> 2;
    long long tid = (long long)blockIdx.x * blockDim.x + threadIdx.x;
    if (tid >= (long long)E * chunk) return;
    int e = (int)(tid / chunk);
    int c = (int)(tid % chunk) << 2;
    int r  = rowI[e];
    int cl = colI[e];
    float w = dinv[r] * dinv[cl];
    float4 v = *(const float4*)(T + (size_t)r * d + c);
    float* dst = agg + (size_t)cl * d + c;
    atomicAdd(dst + 0, v.x * w);
    atomicAdd(dst + 1, v.y * w);
    atomicAdd(dst + 2, v.z * w);
    atomicAdd(dst + 3, v.w * w);
}

__global__ void bias_relu(float* __restrict__ h, const float* __restrict__ b,
                          int M, int d)
{
    long long tid = (long long)blockIdx.x * blockDim.x + threadIdx.x;
    if (tid >= (long long)M * d) return;
    int f = (int)(tid % d);
    h[tid] = fmaxf(h[tid] + b[f], 0.0f);
}

__global__ void cnt_accum(const int* __restrict__ batch, float* __restrict__ cnt, int M) {
    int i = blockIdx.x * blockDim.x + threadIdx.x;
    if (i < M) atomicAdd(&cnt[batch[i]], 1.0f);
}

// pooled[g][0:512] += h ; pooled[g][512+f] = max (features are >=0 post-ReLU,
// so integer atomicMax on the bit pattern is exact; 0-init == empty-graph 0)
__global__ void pool_accum(const float* __restrict__ h, const int* __restrict__ batch,
                           float* __restrict__ pooled, int M)
{
    long long tid = (long long)blockIdx.x * blockDim.x + threadIdx.x;
    if (tid >= (long long)M * 512) return;
    int i = (int)(tid >> 9);
    int f = (int)(tid & 511);
    int g = batch[i];
    float v = h[tid];
    atomicAdd(&pooled[(size_t)g * 1024 + f], v);
    atomicMax((unsigned int*)&pooled[(size_t)g * 1024 + 512 + f], __float_as_uint(v));
}

__global__ void pool_finalize(float* __restrict__ pooled, const float* __restrict__ cnt, int G) {
    long long tid = (long long)blockIdx.x * blockDim.x + threadIdx.x;
    if (tid >= (long long)G * 512) return;
    int g = (int)(tid / 512);
    int f = (int)(tid % 512);
    pooled[(size_t)g * 1024 + f] /= fmaxf(cnt[g], 1.0f);
}

__global__ void concat2(const float* __restrict__ a, const float* __restrict__ b,
                        float* __restrict__ out, int G)
{
    long long tid = (long long)blockIdx.x * blockDim.x + threadIdx.x;
    if (tid >= (long long)G * 256) return;
    int g = (int)(tid / 256);
    int f = (int)(tid % 256);
    out[tid] = (f < 128) ? a[(size_t)g * 128 + f] : b[(size_t)g * 128 + (f - 128)];
}

// out[g] = dot(h[g,:], W) + b   — one wave32 per graph, shuffle reduction
__global__ void final_out(const float* __restrict__ h, const float* __restrict__ W,
                          const float* __restrict__ b, float* __restrict__ out,
                          int G, int K)
{
    int g    = blockIdx.x * (blockDim.x >> 5) + (threadIdx.x >> 5);
    int lane = threadIdx.x & 31;
    if (g >= G) return;
    float s = 0.0f;
    for (int f = lane; f < K; f += 32) s += h[(size_t)g * K + f] * W[f];
    for (int off = 16; off; off >>= 1) s += __shfl_down(s, off, 32);
    if (lane == 0) out[g] = s + b[0];
}

// ---------------------------------------------------------------------------
extern "C" void kernel_launch(void* const* d_in, const int* in_sizes, int n_in,
                              void* d_out, int out_size, void* d_ws, size_t ws_size,
                              hipStream_t stream)
{
    (void)in_sizes; (void)n_in; (void)out_size; (void)ws_size;

    const float* x     = (const float*)d_in[0];
    const int*   ei    = (const int*)d_in[1];   // [2, NE]
    const int*   batch = (const int*)d_in[3];
    const float* sf    = (const float*)d_in[4];
    const float* P[26];
    for (int i = 0; i < 26; ++i) P[i] = (const float*)d_in[5 + i];
    const float *gcn1W=P[0],*gcn1b=P[1],*gcn2W=P[2],*gcn2b=P[3],*gcn3W=P[4],*gcn3b=P[5];
    const float *fcg1W=P[6],*fcg1b=P[7],*fcg2W=P[8],*fcg2b=P[9];
    const float *sg1W=P[10],*sg1b=P[11],*sg2W=P[12],*sg2b=P[13],*sg3W=P[14],*sg3b=P[15];
    const float *fcs1W=P[16],*fcs1b=P[17],*fcs2W=P[18],*fcs2b=P[19];
    const float *fc1W=P[20],*fc1b=P[21],*fc2W=P[22],*fc2b=P[23],*outW=P[24],*outb=P[25];

    // workspace carve-out (aligned to 256B)
    char* wp = (char*)d_ws;
    auto take = [&](size_t bytes) -> void* {
        void* r = (void*)wp; wp += (bytes + 255) & ~(size_t)255; return r;
    };
    float* dinv    = (float*)take((size_t)NN * 4);
    float* bufA    = (float*)take((size_t)NN * 512 * 4);  // agg / layer input
    float* bufT    = (float*)take((size_t)NN * 512 * 4);  // transformed features
    float* pooledX = (float*)take((size_t)NG * 1024 * 4);
    float* pooledS = (float*)take((size_t)NG * 1024 * 4);
    float* cnt     = (float*)take((size_t)NG * 4);
    float* fcTmp   = (float*)take((size_t)NG * 1024 * 4);
    float* gfeatX  = (float*)take((size_t)NG * 128 * 4);
    float* gfeatS  = (float*)take((size_t)NG * 128 * 4);
    float* catbuf  = (float*)take((size_t)NG * 256 * 4);
    float* fc2out  = (float*)take((size_t)NG * 512 * 4);

    const int* rowI = ei;        // edge_index[0] : source
    const int* colI = ei + NE;   // edge_index[1] : target

    auto cdiv = [](long long a, long long b) { return (unsigned)((a + b - 1) / b); };

    // ---- symmetric normalization: dinv = deg^{-1/2}, deg counts targets + self loop
    fill_f32<<<cdiv(NN,256),256,0,stream>>>(dinv, 1.0f, NN);
    deg_accum<<<cdiv(NE,256),256,0,stream>>>(colI, dinv, NE);
    rsqrt_ip <<<cdiv(NN,256),256,0,stream>>>(dinv, NN);

    auto gemm = [&](const float* A, const float* W, const float* bias, float* C,
                    int M, int K, int N, int relu) {
        dim3 grid(N / 16, cdiv(M, 256));      // 8 waves/block, 32 rows/wave
        gemm_wmma_bf16<<<grid, 256, 0, stream>>>(A, W, bias, C, M, K, N, relu);
    };

    auto conv = [&](const float* H, int dIn, const float* W, const float* b, int dOut) {
        gemm(H, W, nullptr, bufT, NN, dIn, dOut, 0);                    // T = H @ W
        self_loop_init<<<cdiv((long long)NN*(dOut/4),256),256,0,stream>>>(bufT, dinv, bufA, NN, dOut);
        edge_scatter  <<<cdiv((long long)NE*(dOut/4),256),256,0,stream>>>(bufT, rowI, colI, dinv, bufA, NE, dOut);
        bias_relu     <<<cdiv((long long)NN*dOut,256),256,0,stream>>>(bufA, b, NN, dOut);
    };

    auto branch = [&](const float* H0,
                      const float* W1, const float* b1, const float* W2, const float* b2,
                      const float* W3, const float* b3,
                      const float* fAW, const float* fAb, const float* fBW, const float* fBb,
                      float* pooled, float* gfeat) {
        conv(H0,   128, W1, b1, 128);
        conv(bufA, 128, W2, b2, 256);
        conv(bufA, 256, W3, b3, 512);
        fill_f32 <<<cdiv((long long)NG*1024,256),256,0,stream>>>(pooled, 0.0f, (long long)NG*1024);
        fill_f32 <<<cdiv(NG,256),256,0,stream>>>(cnt, 0.0f, NG);
        cnt_accum<<<cdiv(NN,256),256,0,stream>>>(batch, cnt, NN);
        pool_accum   <<<cdiv((long long)NN*512,256),256,0,stream>>>(bufA, batch, pooled, NN);
        pool_finalize<<<cdiv((long long)NG*512,256),256,0,stream>>>(pooled, cnt, NG);
        gemm(pooled, fAW, fAb, fcTmp, NG, 1024, 1024, 1);
        gemm(fcTmp,  fBW, fBb, gfeat, NG, 1024, 128,  1);
    };

    branch(x,  gcn1W,gcn1b, gcn2W,gcn2b, gcn3W,gcn3b, fcg1W,fcg1b, fcg2W,fcg2b, pooledX, gfeatX);
    branch(sf, sg1W,sg1b,   sg2W,sg2b,   sg3W,sg3b,   fcs1W,fcs1b, fcs2W,fcs2b, pooledS, gfeatS);

    concat2<<<cdiv((long long)NG*256,256),256,0,stream>>>(gfeatX, gfeatS, catbuf, NG);
    gemm(catbuf, fc1W, fc1b, fcTmp,  NG, 256,  1024, 1);
    gemm(fcTmp,  fc2W, fc2b, fc2out, NG, 1024, 512,  1);
    final_out<<<cdiv(NG,8),256,0,stream>>>(fc2out, outW, outb, (float*)d_out, NG, 512);
}
